// CNN_52604759441677
// MI455X (gfx1250) — compile-verified, matrix operand order checked by reading, not run on hardware
//
#include <hip/hip_runtime.h>
#include <hip/hip_bf16.h>

// ---- sizes from the reference ----
#define S_ 96
#define B_ 8
#define W_ 15
#define EMB_ 300
#define CH_EMB_ 30
#define FILT_ 5
#define CH_K_ 3
#define CH_OUT_ 150          // CH_EMB*FILT
#define POS_D_ 25
#define C_ 64
#define N_K_ 3
#define OUT_ 20
#define DTOT_ (EMB_ + CH_OUT_ + POS_D_)   // 475
#define NSAMP_ (S_ * B_)                  // 768
#define YROWS_ 112                        // 96 padded to 112 (>= 47+64)
#define KK_ 4096                          // 64*64 flattened conv kernel
#define HOUT_ 33                          // 96-64+1
#define YBYTES_ (YROWS_ * C_ * 2)         // 14336 bytes per sample
#define YUNITS8_ (YBYTES_ / 8)            // 1792 8-byte units

typedef __attribute__((ext_vector_type(16))) _Float16 v16h;
typedef __attribute__((ext_vector_type(8)))  _Float16 v8h;
typedef __attribute__((ext_vector_type(8)))  float    v8f;
typedef __attribute__((ext_vector_type(4)))  unsigned int v4u;
typedef __attribute__((ext_vector_type(8)))  int v8i;
typedef __attribute__((ext_vector_type(4)))  int v4i;

union V16 { v16h v; v8h h[2]; };

// ---------- stage 1: char depthwise conv + max-pool ----------
__global__ void k_charfeat(const int* __restrict__ chars, const float* __restrict__ cemb,
                           const float* __restrict__ cw, const float* __restrict__ cb,
                           float* __restrict__ cfeat) {
    int tid = blockIdx.x * blockDim.x + threadIdx.x;
    if (tid >= B_ * S_ * CH_OUT_) return;
    int co = tid % CH_OUT_;
    int ns = tid / CH_OUT_;            // ns = b*S + s
    int ci = co / FILT_;               // depthwise group -> input channel
    const int* ch = chars + ns * W_;
    float w0 = cw[co * 3 + 0], w1 = cw[co * 3 + 1], w2 = cw[co * 3 + 2];
    float bias = cb[co];
    float e[W_];
#pragma unroll
    for (int p = 0; p < W_; ++p) e[p] = cemb[ch[p] * CH_EMB_ + ci];
    float m = -3.4e38f;
#pragma unroll
    for (int p = 0; p < W_ - CH_K_ + 1; ++p) {
        float v = e[p] * w0 + e[p + 1] * w1 + e[p + 2] * w2 + bias;
        m = fmaxf(m, v);
    }
    cfeat[tid] = m;                    // (B,S,150)
}

// ---------- stage 2: base[b,s,c] = [word_emb ; char_feat] . Ww[c] ----------
__global__ void k_base(const int* __restrict__ words, const float* __restrict__ wemb,
                       const float* __restrict__ cfeat, const float* __restrict__ lin_w,
                       float* __restrict__ basep) {
    int tid = blockIdx.x * blockDim.x + threadIdx.x;
    if (tid >= B_ * S_ * C_) return;
    int c = tid & 63;
    int bs = tid >> 6;                 // b*S + s
    int s = bs % S_;
    int b = bs / S_;
    const float* wr = wemb + (size_t)words[s * B_ + b] * EMB_;   // words is (S,B)
    const float* lw = lin_w + c * DTOT_;
    float acc = 0.f;
    for (int d = 0; d < EMB_; ++d) acc += wr[d] * lw[d];
    const float* cf = cfeat + bs * CH_OUT_;
    for (int d = 0; d < CH_OUT_; ++d) acc += cf[d] * lw[EMB_ + d];
    basep[tid] = acc;                  // (B,S,64)
}

// ---------- stage 3: pos_part[i,s,c] = pos_emb[|i-s|] . Wp[c] ----------
__global__ void k_pos(const float* __restrict__ pemb, const float* __restrict__ lin_w,
                      float* __restrict__ posp) {
    int tid = blockIdx.x * blockDim.x + threadIdx.x;
    if (tid >= S_ * S_ * C_) return;
    int c = tid & 63;
    int is = tid >> 6;                 // i*S + s
    int s = is % S_;
    int i = is / S_;
    int d = (i > s) ? (i - s) : (s - i);
    const float* pr = pemb + d * POS_D_;
    const float* lw = lin_w + c * DTOT_ + (EMB_ + CH_OUT_);
    float acc = 0.f;
#pragma unroll
    for (int p = 0; p < POS_D_; ++p) acc += pr[p] * lw[p];
    posp[tid] = acc;                   // (S,S,64)
}

// ---------- stage 4: assemble y (f16, padded to 112 rows) ----------
__global__ void k_buildy(const float* __restrict__ basep, const float* __restrict__ posp,
                         const float* __restrict__ lin_b, _Float16* __restrict__ yh) {
    int tid = blockIdx.x * blockDim.x + threadIdx.x;
    if (tid >= NSAMP_ * YROWS_ * C_) return;
    int c = tid & 63;
    int t = tid >> 6;
    int r = t % YROWS_;
    int n = t / YROWS_;                // n = i*B + b
    float v = 0.f;
    if (r < S_) {
        int i = n >> 3;
        int b = n & 7;
        v = basep[((b * S_ + r) << 6) + c] + posp[((i * S_ + r) << 6) + c] + lin_b[c];
    }
    yh[tid] = (_Float16)v;
}

// ---------- stage 5: conv weights f32 -> f16 (flat (3,64,4096)) ----------
__global__ void k_cvtw(const float* __restrict__ cw, _Float16* __restrict__ wh) {
    int tid = blockIdx.x * blockDim.x + threadIdx.x;
    if (tid >= N_K_ * C_ * KK_) return;
    wh[tid] = (_Float16)cw[tid];
}

// ---------- stage 6: big conv-as-GEMM via WMMA, relu+bias, max over h ----------
__global__ void __launch_bounds__(128)
k_conv_wmma(const _Float16* __restrict__ yh, const _Float16* __restrict__ wh,
            const float* __restrict__ convb, float* __restrict__ feats) {
    __shared__ _Float16 ylds[YROWS_ * C_];   // 14336 bytes
    int bx = blockIdx.x;
    int n = bx % NSAMP_;
    int j = bx / NSAMP_;

    // ---- stage the 14336B sample slab into LDS ----
    unsigned lbase = (unsigned)(unsigned long long)(void*)ylds;   // low 32b of shared-aperture ptr == LDS byte offset
    const _Float16* src = yh + (size_t)n * (YROWS_ * C_);
#if __has_builtin(__builtin_amdgcn_tensor_load_to_lds)
    // Tensor Data Mover: one D# descriptor moves the whole tile; issued by wave 0,
    // tracked with TENSORcnt. Descriptor per CDNA5 ISA.
    if (threadIdx.x < 32) {
        unsigned long long ga = (unsigned long long)src;
        v4u g0;
        g0.x = 1u;                                              // count=1 (valid user D#)
        g0.y = lbase;                                           // lds_addr
        g0.z = (unsigned)ga;                                    // global_addr[31:0]
        g0.w = ((unsigned)(ga >> 32) & 0x01FFFFFFu)             // global_addr[56:32]
             | (2u << 30);                                      // type=2 ("image")
        v8i g1 = {};
        g1[0] = (3 << 16);                                      // data_size=3 (8-byte units)
        g1[1] = (int)((YUNITS8_ & 0xFFFF) << 16);               // tensor_dim0[15:0]  -> bits[63:48]
        g1[2] = (int)((YUNITS8_ >> 16) & 0xFFFF)                // tensor_dim0[31:16] -> bits[79:64]
              | (int)(1u << 16);                                // tensor_dim1[15:0]=1 -> bits[95:80]
        g1[3] = (int)((unsigned)YUNITS8_ << 16);                // tile_dim0 -> bits[127:112]
        g1[4] = 1;                                              // tile_dim1=1 -> bits[143:128]
        g1[5] = YUNITS8_;                                       // tensor_dim0_stride[31:0]
        v4i gz = {};
#if defined(__clang_major__) && (__clang_major__ >= 23)
        v8i g5 = {};
        __builtin_amdgcn_tensor_load_to_lds(g0, g1, gz, gz, g5, 0);
#else
        __builtin_amdgcn_tensor_load_to_lds(g0, g1, gz, gz, 0);
#endif
        __builtin_amdgcn_s_wait_tensorcnt(0);
    }
#else
    // Fallback: CDNA5 async memory->LDS loads (ASYNCcnt), no VGPR round-trip.
    for (int t = threadIdx.x; t < (YROWS_ * C_) / 8; t += 128) {
        unsigned long long ga = (unsigned long long)(src + t * 8);
        unsigned la = lbase + t * 16;
        asm volatile("global_load_async_to_lds_b128 %0, %1, off"
                     :: "v"(la), "v"(ga) : "memory");
    }
    asm volatile("s_wait_asynccnt 0x0" ::: "memory");
#endif
    __syncthreads();

    int wave = threadIdx.x >> 5;       // 0..3 -> output-channel tile
    int lane = threadIdx.x & 31;
    int half = lane >> 4;              // 0 or 1 (K sub-group)
    int lr   = lane & 15;              // A row-in-tile / B column
    int klo  = half << 3;              // 0 or 8
    int oc   = (wave << 4) + lr;       // output channel 0..63

    const _Float16* Bcol = wh + ((size_t)(j * C_ + oc)) * KK_;

    v8f acc0 = {}, acc1 = {}, acc2 = {};
    for (int kb = 0; kb < KK_; kb += 32) {
        int kh = kb >> 6;
        int kw0 = kb & 63;
        // prefetch weight stream ~4KB ahead (helps first pass; then L2-resident)
        if (kb + 2048 < KK_) __builtin_prefetch(Bcol + kb + 2048 + klo, 0, 0);
        // B fragment: column oc, K chunk (contiguous in memory)
        V16 bf;
        bf.h[0] = *(const v8h*)(Bcol + kb + klo);
        bf.h[1] = *(const v8h*)(Bcol + kb + 16 + klo);
        // A fragments: rows h = mt*16 + lr, element y[n, h+kh, kw0 + t]
        const _Float16* ar = ylds + (kh + lr) * C_ + kw0;
        V16 a0, a1, a2;
        a0.h[0] = *(const v8h*)(ar + klo);
        a0.h[1] = *(const v8h*)(ar + 16 + klo);
        a1.h[0] = *(const v8h*)(ar + 16 * C_ + klo);
        a1.h[1] = *(const v8h*)(ar + 16 * C_ + 16 + klo);
        a2.h[0] = *(const v8h*)(ar + 32 * C_ + klo);
        a2.h[1] = *(const v8h*)(ar + 32 * C_ + 16 + klo);
        acc0 = __builtin_amdgcn_wmma_f32_16x16x32_f16(false, a0.v, false, bf.v, (short)0, acc0, false, false);
        acc1 = __builtin_amdgcn_wmma_f32_16x16x32_f16(false, a1.v, false, bf.v, (short)0, acc1, false, false);
        acc2 = __builtin_amdgcn_wmma_f32_16x16x32_f16(false, a2.v, false, bf.v, (short)0, acc2, false, false);
    }

    // bias + relu + max over valid h (C layout: lane holds h = mt*16 + r (+8 if half))
    float bias = convb[j * C_ + oc];
    float m = 0.0f;                    // relu makes the max >= 0, so 0-init == relu-then-max
    int hbase = half * 8;
#pragma unroll
    for (int r = 0; r < 8; ++r) {
        int h0 = hbase + r;
        { float v = acc0[r] + bias; if (h0      < HOUT_) m = fmaxf(m, v); }
        { float v = acc1[r] + bias; if (h0 + 16 < HOUT_) m = fmaxf(m, v); }
        { float v = acc2[r] + bias; if (h0 + 32 < HOUT_) m = fmaxf(m, v); }
    }
    // lanes L and L+16 hold complementary h sets for the same oc
    float other = __shfl_xor(m, 16, 32);
    m = fmaxf(m, other);
    if (half == 0) feats[n * (N_K_ * C_) + j * C_ + oc] = m;
}

// ---------- stage 7: final FC (S,B,20) ----------
__global__ void k_fc(const float* __restrict__ feats, const float* __restrict__ fcw,
                     const float* __restrict__ fcb, float* __restrict__ out) {
    int tid = blockIdx.x * blockDim.x + threadIdx.x;
    if (tid >= S_ * B_ * OUT_) return;
    int o = tid % OUT_;
    int sb = tid / OUT_;               // s*B + b == feats row n (n = i*B + b, i == s)
    const float* f = feats + sb * (N_K_ * C_);
    const float* w = fcw + o * (N_K_ * C_);
    float acc = fcb[o];
#pragma unroll 4
    for (int d = 0; d < N_K_ * C_; ++d) acc += f[d] * w[d];
    out[tid] = acc;
}

extern "C" void kernel_launch(void* const* d_in, const int* in_sizes, int n_in,
                              void* d_out, int out_size, void* d_ws, size_t ws_size,
                              hipStream_t stream) {
    const int*   words      = (const int*)d_in[0];
    const int*   chars      = (const int*)d_in[1];
    const float* word_emb   = (const float*)d_in[2];
    const float* char_emb   = (const float*)d_in[3];
    const float* char_cnn_w = (const float*)d_in[4];
    const float* char_cnn_b = (const float*)d_in[5];
    const float* pos_emb    = (const float*)d_in[6];
    const float* lin_w      = (const float*)d_in[7];
    const float* lin_b      = (const float*)d_in[8];
    const float* conv_w     = (const float*)d_in[9];
    const float* conv_b     = (const float*)d_in[10];
    const float* fc_w       = (const float*)d_in[11];
    const float* fc_b       = (const float*)d_in[12];
    float* out = (float*)d_out;

    // workspace layout (bytes, all 16B-aligned)
    char* ws = (char*)d_ws;
    float*    cfeat = (float*)(ws + 0);                  //  768*150*4 = 460800
    float*    basep = (float*)(ws + 460800);             //  768*64*4  = 196608
    float*    posp  = (float*)(ws + 657408);             // 9216*64*4  = 2359296
    _Float16* yh    = (_Float16*)(ws + 3016704);         //  768*112*64*2 = 11010048
    _Float16* wh    = (_Float16*)(ws + 14026752);        //  786432*2  = 1572864
    float*    feats = (float*)(ws + 15599616);           //  768*192*4 = 589824

    k_charfeat<<<(B_ * S_ * CH_OUT_ + 255) / 256, 256, 0, stream>>>(chars, char_emb, char_cnn_w, char_cnn_b, cfeat);
    k_base<<<(B_ * S_ * C_ + 255) / 256, 256, 0, stream>>>(words, word_emb, cfeat, lin_w, basep);
    k_pos<<<(S_ * S_ * C_ + 255) / 256, 256, 0, stream>>>(pos_emb, lin_w, posp);
    k_buildy<<<(NSAMP_ * YROWS_ * C_ + 255) / 256, 256, 0, stream>>>(basep, posp, lin_b, yh);
    k_cvtw<<<(N_K_ * C_ * KK_ + 255) / 256, 256, 0, stream>>>(conv_w, wh);
    k_conv_wmma<<<NSAMP_ * N_K_, 128, 0, stream>>>(yh, wh, conv_b, feats);
    k_fc<<<(S_ * B_ * OUT_ + 255) / 256, 256, 0, stream>>>(feats, fc_w, fc_b, out);
}
